// CRF_33913061769500
// MI455X (gfx1250) — matrix operand order repaired and checked by reading, not compiled, not run on hardware
//
#include <hip/hip_runtime.h>
#include <hip/hip_fp16.h>

// CRF neg-log-likelihood for MI455X (gfx1250, wave32).
// The log-semiring scan step
//     alpha'[b,j] = logsumexp_i(alpha[b,i] + trans[i,j]) + emit[b,t,j]
// is computed exactly as  c[b] + log((P @ E)[b,j]) + emit, where
//     c = rowmax(alpha), P = exp(alpha - c) (fp16), E = exp(trans) (fp16, const),
// so the T x T contraction runs on v_wmma_f32_16x16x32_f16 with E resident in VGPRs.

typedef __attribute__((ext_vector_type(16))) _Float16 v16h_t;
typedef __attribute__((ext_vector_type(8)))  _Float16 v8h_t;
typedef __attribute__((ext_vector_type(2)))  __fp16   fp16x2_t;   // cvt_pkrtz result type
typedef __attribute__((ext_vector_type(8)))  float    v8f_t;
typedef __attribute__((ext_vector_type(4)))  float    v4f_t;

namespace {
constexpr int kB  = 128;
constexpr int kS  = 256;
constexpr int kT  = 52;        // TAGSET + START + STOP
constexpr int kTP = 64;        // padded tag dim (multiple of 32 for WMMA K)
constexpr int kSTART = kT - 2; // 50
constexpr int kSTOP  = kT - 1; // 51
constexpr float kNEG   = -1.0e30f;
constexpr float kLOG2E = 1.44269504088896340736f;
constexpr float kLN2   = 0.69314718055994530942f;
}

union AFrag { v16h_t v; _Float16 h[16]; };
union CFrag { v8f_t  v; float    f[8];  };
union PPack { fp16x2_t h2[4]; v8h_t v; };

__global__ __launch_bounds__(128)
void crf_forward_kernel(const float* __restrict__ feats,
                        const float* __restrict__ trans,
                        const unsigned char* __restrict__ mask,
                        float* __restrict__ fwd_out)
{
    __shared__ float        sAlpha[16][kTP];   // running log-alpha (16-batch M tile)
    __shared__ _Float16     sP[16][kTP];       // exp(alpha - c), WMMA A source
    __shared__ _Float16     sE[kTP][kTP];      // exp(trans), WMMA B source
    __shared__ float        sC[16];            // per-row max
    __shared__ unsigned int sMask[kS];         // bit m = mask[b0+m][t]

    const int tid  = threadIdx.x;
    const int b0   = blockIdx.x * 16;       // batch tile base
    const int wave = tid >> 5;              // 0..3 -> N tile
    const int lane = tid & 31;
    const int ln15 = lane & 15;
    const int hi   = lane >> 4;             // half-wave select

    // ---- E = exp(trans), zero-padded to 64x64 (exp(-10000) underflows to 0) ----
    for (int idx = tid; idx < kTP * kTP; idx += 128) {
        const int i = idx >> 6, j = idx & 63;
        float v = 0.0f;
        if (i < kT && j < kT)
            v = __builtin_amdgcn_exp2f(trans[i * kT + j] * kLOG2E);
        sE[i][j] = (_Float16)v;
    }
    // ---- alpha init: feats[:,0,:] + trans[START,:] ----
    for (int idx = tid; idx < 16 * kTP; idx += 128) {
        const int m = idx >> 6, j = idx & 63;
        float v = kNEG;
        if (j < kT)
            v = fmaxf(feats[((b0 + m) * kS + 0) * kT + j] + trans[kSTART * kT + j], kNEG);
        sAlpha[m][j] = v;
    }
    // ---- pack mask tile into per-step 16-bit masks ----
    for (int t = tid; t < kS; t += 128) {
        unsigned int mb = 0;
        #pragma unroll
        for (int m = 0; m < 16; ++m)
            mb |= (mask[(b0 + m) * kS + t] != 0 ? 1u : 0u) << m;
        sMask[t] = mb;
    }
    __syncthreads();

    // ---- B fragments (E is scan-invariant; keep resident in VGPRs) ----
    // 16-bit B 32x16 layout: half h holds K = kt*32 + (h < 8 ? 8*hi + h : 16 + 8*hi + h - 8), N = 16*wave + ln15.
    AFrag bf[2];
    #pragma unroll
    for (int kt = 0; kt < 2; ++kt)
        #pragma unroll
        for (int h = 0; h < 16; ++h)
            bf[kt].h[h] = sE[kt * 32 + (h < 8 ? hi * 8 + h : 16 + hi * 8 + (h - 8))][wave * 16 + ln15];

    const int mrow  = tid >> 3;  // 0..15: row this thread reduces
    const int chunk = tid & 7;   // 0..7 : 8-column chunk
    const int N     = wave * 16 + ln15;          // output column of this thread
    const int Nc    = (N < kT) ? N : 0;          // clamped (branchless epilogue)
    const bool Nvalid = (N < kT);

    for (int t = 1; t <= kS; ++t) {
        __syncthreads();                       // alpha stable; sP consumed

        // ---- early-issue global traffic for this step (hidden behind max/exp/WMMA) ----
        float emit[8];
        unsigned int mbits = 0;
        if (t < kS) {
            #pragma unroll
            for (int r = 0; r < 8; ++r)
                emit[r] = feats[((b0 + (r + hi * 8)) * kS + t) * kT + Nc];
            mbits = sMask[t];
            if (t + 1 < kS)
                __builtin_prefetch(&feats[((b0 + hi * 8) * kS + (t + 1)) * kT + Nc], 0, 1);
        }

        // ---- c = rowmax(alpha): 2x ds_load_b128 + in-wave shfl_xor tree ----
        const v4f_t a0 = *reinterpret_cast<const v4f_t*>(&sAlpha[mrow][chunk * 8]);
        const v4f_t a1 = *reinterpret_cast<const v4f_t*>(&sAlpha[mrow][chunk * 8 + 4]);
        float mx = fmaxf(fmaxf(fmaxf(a0.x, a0.y), fmaxf(a0.z, a0.w)),
                         fmaxf(fmaxf(a1.x, a1.y), fmaxf(a1.z, a1.w)));
        mx = fmaxf(mx, __shfl_xor(mx, 1));
        mx = fmaxf(mx, __shfl_xor(mx, 2));
        mx = fmaxf(mx, __shfl_xor(mx, 4));     // all 8 row-threads hold row max
        if (chunk == 0) sC[mrow] = mx;

        // ---- P = exp2((alpha - c) * log2e), packed fp16 store (ds_store_b128) ----
        PPack pk;
        pk.h2[0] = __builtin_amdgcn_cvt_pkrtz(__builtin_amdgcn_exp2f((a0.x - mx) * kLOG2E),
                                              __builtin_amdgcn_exp2f((a0.y - mx) * kLOG2E));
        pk.h2[1] = __builtin_amdgcn_cvt_pkrtz(__builtin_amdgcn_exp2f((a0.z - mx) * kLOG2E),
                                              __builtin_amdgcn_exp2f((a0.w - mx) * kLOG2E));
        pk.h2[2] = __builtin_amdgcn_cvt_pkrtz(__builtin_amdgcn_exp2f((a1.x - mx) * kLOG2E),
                                              __builtin_amdgcn_exp2f((a1.y - mx) * kLOG2E));
        pk.h2[3] = __builtin_amdgcn_cvt_pkrtz(__builtin_amdgcn_exp2f((a1.z - mx) * kLOG2E),
                                              __builtin_amdgcn_exp2f((a1.w - mx) * kLOG2E));
        *reinterpret_cast<v8h_t*>(&sP[mrow][chunk * 8]) = pk.v;
        __syncthreads();                       // P + sC ready

        if (t < kS) {
            // ---- A fragments: 16-bit A 16x32 layout (ISA 7.12.2) ----
            AFrag af[2];
            #pragma unroll
            for (int kt = 0; kt < 2; ++kt) {
                #pragma unroll
                for (int v = 0; v < 8; ++v) {
                    #pragma unroll
                    for (int p = 0; p < 2; ++p) {
                        const int kk = (v < 4)
                            ? (hi ? 8 + 2 * v + p : 2 * v + p)
                            : (hi ? 24 + 2 * (v - 4) + p : 16 + 2 * (v - 4) + p);
                        af[kt].h[2 * v + p] = sP[ln15][kt * 32 + kk];
                    }
                }
            }

            // ---- acc = P @ E  (16x16 N-tile per wave, K = 64 via two WMMAs) ----
            CFrag acc = {};
            acc.v = __builtin_amdgcn_wmma_f32_16x16x32_f16(
                false, af[0].v, false, bf[0].v, (short)0, acc.v, false, false);
            acc.v = __builtin_amdgcn_wmma_f32_16x16x32_f16(
                false, af[1].v, false, bf[1].v, (short)0, acc.v, false, false);

            // ---- alpha' = c + log(acc) + emit, masked keep-old (branchless) ----
            // C/D layout: VGPR r -> M = r + 8*hi, N = 16*wave + ln15 (unique thread/elem)
            #pragma unroll
            for (int r = 0; r < 8; ++r) {
                const int M = r + hi * 8;
                const float oldv = sAlpha[M][N];
                const float lacc = __builtin_amdgcn_logf(acc.f[r]) * kLN2; // ln(acc)
                float nv = fmaxf(sC[M] + lacc + emit[r], kNEG);
                nv = Nvalid ? nv : kNEG;
                const bool mm = ((mbits >> M) & 1u) != 0u;
                sAlpha[M][N] = mm ? nv : oldv;
            }
        } else {
            // ---- termination: logsumexp(alpha + trans[:,STOP]) = c + log(P . E[:,STOP]) ----
            if (tid < 16) {
                float s = 0.0f;
                #pragma unroll
                for (int k = 0; k < kTP; ++k)
                    s += (float)sP[tid][k] * (float)sE[k][kSTOP];
                fwd_out[b0 + tid] = sC[tid] + __builtin_amdgcn_logf(s) * kLN2;
            }
        }
    }
}

__global__ __launch_bounds__(128)
void crf_gold_kernel(const float* __restrict__ feats,
                     const float* __restrict__ trans,
                     const unsigned char* __restrict__ mask,
                     const int* __restrict__ tags,
                     float* __restrict__ gold_out)
{
    const int b = threadIdx.x;
    if (b >= kB) return;
    float sum = 0.0f;
    int prev = kSTART;
    int last = kSTART;
    for (int t = 0; t < kS; ++t) {
        if (mask[b * kS + t]) {
            const int tag = tags[b * kS + t];
            sum += feats[(b * kS + t) * kT + tag] + trans[prev * kT + tag];
            prev = tag;
            last = tag;
        }
    }
    sum += trans[last * kT + kSTOP];
    gold_out[b] = sum;
}

// Deterministic ordered reduction (no float atomics -> bitwise-stable replays).
__global__ void crf_reduce_kernel(const float* __restrict__ fwd,
                                  const float* __restrict__ gold,
                                  float* __restrict__ out)
{
    if (threadIdx.x == 0 && blockIdx.x == 0) {
        float acc = 0.0f;
        for (int b = 0; b < kB; ++b) acc += fwd[b];
        for (int b = 0; b < kB; ++b) acc -= gold[b];
        out[0] = acc;
    }
}

extern "C" void kernel_launch(void* const* d_in, const int* in_sizes, int n_in,
                              void* d_out, int out_size, void* d_ws, size_t ws_size,
                              hipStream_t stream) {
    const float*         feats = (const float*)d_in[0];          // (B,S,T) f32
    const float*         trans = (const float*)d_in[1];          // (T,T)   f32
    const unsigned char* mask  = (const unsigned char*)d_in[2];  // (B,S)   bool
    const int*           tags  = (const int*)d_in[3];            // (B,S)   i32

    float* fwd  = (float*)d_ws;        // [B] per-batch log Z
    float* gold = fwd + kB;            // [B] per-batch gold score

    crf_forward_kernel<<<kB / 16, 128, 0, stream>>>(feats, trans, mask, fwd);
    crf_gold_kernel<<<1, 128, 0, stream>>>(feats, trans, mask, tags, gold);
    crf_reduce_kernel<<<1, 32, 0, stream>>>(fwd, gold, (float*)d_out);
}
